// VQVAE_31808527794306
// MI455X (gfx1250) — matrix operand order repaired and checked by reading, not compiled, not run on hardware
//
#include <hip/hip_runtime.h>
#include <cstdint>

typedef __attribute__((ext_vector_type(16))) __bf16    v16bf;
typedef __attribute__((ext_vector_type(8)))  float     v8f;
typedef __attribute__((ext_vector_type(4)))  uint32_t  v4u;
typedef __attribute__((ext_vector_type(8)))  uint32_t  v8u;

#define NBLK 4   // 4 N-subtiles (16x64 output per wave) -> reuse A fragment, 4 WMMAs/K-step

union AFrag { v16bf v; uint16_t u[16]; uint32_t d[8]; };

__device__ __forceinline__ uint16_t f2bf(float f) {
    union { float f; uint32_t u; } x; x.f = f;
    uint32_t u = x.u;
    uint32_t r = u + 0x7FFFu + ((u >> 16) & 1u);   // round-to-nearest-even
    return (uint16_t)(r >> 16);
}

__device__ __forceinline__ uint32_t ordkey(float f) {
    union { float f; uint32_t u; } x; x.f = f;
    uint32_t u = x.u;
    return (u & 0x80000000u) ? ~u : (u | 0x80000000u);   // monotone float->uint map
}

// ---------------- cast / layout kernels ----------------

__global__ void cast_f32_bf16(const float* __restrict__ in, uint16_t* __restrict__ out, long n) {
    long i = (long)blockIdx.x * blockDim.x + threadIdx.x;
    long stride = (long)gridDim.x * blockDim.x;
    for (; i < n; i += stride) out[i] = f2bf(in[i]);
}

// torch ConvTranspose2d weight (Cin,Cout,4,4) -> equivalent-conv bf16 [Cout][Cin*16],
// K index = cin*16 + kh*4 + kw, taps spatially flipped.
__global__ void permute_dec_w(const float* __restrict__ in, uint16_t* __restrict__ out,
                              int Cin, int Cout) {
    int K = Cin * 16;
    long total = (long)Cout * K;
    long i = (long)blockIdx.x * blockDim.x + threadIdx.x;
    if (i >= total) return;
    int m = (int)(i / K);
    int rem = (int)(i % K);
    int cin = rem >> 4, kh = (rem >> 2) & 3, kw = rem & 3;
    long src = (((long)cin * Cout + m) * 4 + (3 - kh)) * 4 + (3 - kw);
    out[i] = f2bf(in[src]);
}

__global__ void cnorm_kernel(const float* __restrict__ embed, float* __restrict__ cnorm) {
    int m = blockIdx.x * blockDim.x + threadIdx.x;
    if (m >= 1024) return;
    float s = 0.f;
    const float* row = embed + (long)m * 512;
    for (int j = 0; j < 512; ++j) s += row[j] * row[j];
    cnorm[m] = s;
}

__global__ void init_keys(unsigned long long* keys, int n) {
    int i = blockIdx.x * blockDim.x + threadIdx.x;
    if (i < n) keys[i] = ~0ull;
}

// ---------------- implicit-GEMM conv (WMMA bf16 + TDM weight staging) ----------------
// D[Cout][N] = W[Cout][Cin*16] x im2col[Cin*16][N], N = B*Ho*Wo.
// Block = 8 waves sharing one 16-row weight block (m0); waves fan out over N.
// Per K-chunk, wave 0 DMA's W[m0:m0+16][kc0:kc0+KC] into LDS with tensor_load_to_lds
// (TDM zero-fills beyond tensor dims -> no A-side edge guards needed).
__global__ void conv_wmma_kernel(const uint16_t* __restrict__ act,
                                 const uint16_t* __restrict__ wgt,
                                 const float* __restrict__ bias,
                                 uint16_t* __restrict__ out_bf,
                                 float* __restrict__ out_f32,
                                 int B, int Cin, int Hin, int Win,
                                 int Cout, int Ho, int Wo,
                                 int dilated, int final_tanh) {
    extern __shared__ uint16_t smem[];          // 16 x KC bf16 weight tile (dyn LDS, base 0)
    const int K    = Cin * 16;
    const int Kpad = (K + 31) & ~31;
    const int KC   = Kpad < 1024 ? Kpad : 1024; // chunk: <=32KB LDS
    const int HoWo = Ho * Wo;
    const int Ntot = B * HoWo;
    const int tm   = (Cout + 15) >> 4;
    const int tnw  = (Ntot + 63) >> 6;          // 16x64 wave tiles along N

    const int mtb    = blockIdx.x % tm;
    const int ng     = blockIdx.x / tm;
    const int waveid = threadIdx.x >> 5;
    const int ntw    = ng * 8 + waveid;
    const int m0     = mtb * 16;
    const int n0     = ntw * 64;
    const bool wactive = ntw < tnw;             // inactive waves still run barriers

    const int lane  = threadIdx.x & 31;
    const int lrow  = lane & 15;     // A: M index; B/C/D: N column
    const int lhalf = lane >> 4;
    const int kb    = lhalf * 8;

    // decode the NBLK output pixels this lane owns
    int bb[NBLK], hh[NBLK], ww[NBLK];
    bool nvalid[NBLK];
    for (int t = 0; t < NBLK; ++t) {
        int nn = n0 + t * 16 + lrow;
        nvalid[t] = wactive && (nn < Ntot);
        int nc = (nn < Ntot) ? nn : 0;
        bb[t] = nc / HoWo;
        int rem = nc % HoWo;
        hh[t] = rem / Wo;
        ww[t] = rem % Wo;
    }

    v8f c[NBLK];
    for (int t = 0; t < NBLK; ++t)
        for (int i = 0; i < 8; ++i) c[t][i] = 0.f;

    for (int kc0 = 0; kc0 < K; kc0 += KC) {
        __syncthreads();                        // previous chunk fully consumed
        if (threadIdx.x < 32) {
            // ---- Tensor DMA descriptor: 2D tile 16 x KC of W[Cout][K] ----
            unsigned long long ga = (unsigned long long)(uintptr_t)wgt
                                  + ((unsigned long long)m0 * (unsigned)K + (unsigned)kc0) * 2ull;
            uint32_t td0 = (uint32_t)(K - kc0);     // cols remaining (tile-relative OOB)
            uint32_t td1 = (uint32_t)(Cout - m0);   // rows remaining
            v4u g0;
            g0[0] = 1u;                                              // count=1, load, no gather
            g0[1] = 0u;                                              // lds_addr = 0 (dyn LDS)
            g0[2] = (uint32_t)ga;                                    // global_addr[31:0]
            g0[3] = ((uint32_t)(ga >> 32) & 0x01FFFFFFu) | 0x80000000u; // addr[56:32] | type=2
            v8u g1;
            g1[0] = 0x00010000u;                                     // data_size = 2 bytes
            g1[1] = (td0 & 0xFFFFu) << 16;                           // tensor_dim0[15:0]
            g1[2] = (td0 >> 16) | ((td1 & 0xFFFFu) << 16);           // dim0 hi | dim1 lo
            g1[3] = (td1 >> 16) | ((uint32_t)KC << 16);              // dim1 hi | tile_dim0=KC
            g1[4] = 16u;                                             // tile_dim1=16, tile_dim2=0
            g1[5] = (uint32_t)K;                                     // tensor_dim0_stride lo
            g1[6] = 0u;                                              // stride hi | dim1_stride lo
            g1[7] = 0u;
            asm volatile("tensor_load_to_lds %0, %1" :: "s"(g0), "s"(g1) : "memory");
            __builtin_amdgcn_s_wait_tensorcnt(0);
        }
        __syncthreads();                        // weight tile visible to all waves

        int kend = kc0 + KC;
        if (kend > Kpad) kend = Kpad;
        for (int k0 = kc0; k0 < kend; k0 += 32) {
            // ---- A fragment from LDS (16x32 bf16, ISA layout); TDM zero-padded ----
            AFrag a;
            {
                const uint32_t* s32 = (const uint32_t*)smem;
                int base = lrow * KC - kc0;                 // element offset
                for (int vv = 0; vv < 8; ++vv) {
                    int kk = k0 + kb + ((vv < 4) ? 2 * vv : 16 + 2 * (vv - 4));
                    a.d[vv] = s32[(base + kk) >> 1];
                }
            }
            // ---- B fragments: 32x16 bf16 im2col gathers ----
            AFrag bf[NBLK];
            for (int t = 0; t < NBLK; ++t) {
                for (int vv = 0; vv < 8; ++vv) {
                    for (int e = 0; e < 2; ++e) {
                        int kc = k0 + lhalf * 8 + ((vv < 4) ? 2 * vv : 16 + 2 * (vv - 4)) + e;
                        uint16_t val = 0;
                        if (nvalid[t] && kc < K) {
                            int cin = kc >> 4, rr = kc & 15, kh = rr >> 2, kw = rr & 3;
                            if (!dilated) {
                                int hi = 2 * hh[t] + kh - 1, wi = 2 * ww[t] + kw - 1;
                                if ((unsigned)hi < (unsigned)Hin && (unsigned)wi < (unsigned)Win)
                                    val = act[(((size_t)bb[t] * Cin + cin) * Hin + hi) * Win + wi];
                            } else {
                                int hd = hh[t] + kh - 2, wd = ww[t] + kw - 2;
                                if (hd >= 0 && wd >= 0 && !(hd & 1) && !(wd & 1)) {
                                    int hi = hd >> 1, wi = wd >> 1;
                                    if (hi < Hin && wi < Win)
                                        val = act[(((size_t)bb[t] * Cin + cin) * Hin + hi) * Win + wi];
                                }
                            }
                        }
                        bf[t].u[2 * vv + e] = val;
                    }
                }
            }
            // ---- 4 WMMAs, A reused ----
            for (int t = 0; t < NBLK; ++t)
                c[t] = __builtin_amdgcn_wmma_f32_16x16x32_bf16(
                    false, a.v, false, bf[t].v, (short)0, c[t], false, false);
        }
    }

    // ---- epilogue: bias + activation + store (C/D layout: m = lhalf*8+vv, n = lrow) ----
    for (int t = 0; t < NBLK; ++t) {
        if (!nvalid[t]) continue;
        for (int vv = 0; vv < 8; ++vv) {
            int m = m0 + lhalf * 8 + vv;
            if (m >= Cout) continue;
            float val = c[t][vv] + bias[m];
            size_t oidx = (((size_t)bb[t] * Cout + m) * Ho + hh[t]) * Wo + ww[t];
            if (final_tanh) out_f32[oidx] = tanhf(val);
            else            out_bf[oidx]  = f2bf(fmaxf(val, 0.f));
        }
    }
}

// ---------------- quantizer GEMM: scores = E . z^T, argmin via atomic key ----------------
// z: bf16 NCHW (32,512,16,16); eb: bf16 [1024][512]; keys[n] = min over codes.
__global__ void vq_wmma_kernel(const uint16_t* __restrict__ z,
                               const uint16_t* __restrict__ eb,
                               const float* __restrict__ cnorm,
                               unsigned long long* __restrict__ keys) {
    const int tm = 64, tn = 512;                 // M=1024, N=8192, K=512
    int wave = blockIdx.x * (blockDim.x >> 5) + (threadIdx.x >> 5);
    if (wave >= tm * tn) return;
    const int mt = wave % tm, nt = wave / tm;
    const int m0 = mt * 16, n0 = nt * 16;
    const int lane = threadIdx.x & 31, lrow = lane & 15, lhalf = lane >> 4;

    const int n  = n0 + lrow;
    const int b  = n >> 8;
    const int hw = n & 255;

    v8f c;
    for (int i = 0; i < 8; ++i) c[i] = 0.f;

    for (int k0 = 0; k0 < 512; k0 += 32) {
        AFrag a, bf;
        int m = m0 + lrow, kb = lhalf * 8;
        for (int vv = 0; vv < 8; ++vv) {
            int kk = k0 + kb + ((vv < 4) ? 2 * vv : 16 + 2 * (vv - 4));
            uint32_t dw;
            __builtin_memcpy(&dw, eb + (size_t)m * 512 + kk, 4);
            a.d[vv] = dw;
        }
        for (int vv = 0; vv < 8; ++vv) {
            for (int e = 0; e < 2; ++e) {
                int kc = k0 + lhalf * 8 + ((vv < 4) ? 2 * vv : 16 + 2 * (vv - 4)) + e;      // kc = channel d
                bf.u[2 * vv + e] = z[((size_t)b * 512 + kc) * 256 + hw];
            }
        }
        c = __builtin_amdgcn_wmma_f32_16x16x32_bf16(
                false, a.v, false, bf.v, (short)0, c, false, false);
    }

    for (int vv = 0; vv < 8; ++vv) {
        int m = m0 + lhalf * 8 + vv;
        float r = cnorm[m] - 2.0f * c[vv];                 // ||E_m||^2 - 2 z.E_m
        unsigned long long key = ((unsigned long long)ordkey(r) << 32) | (unsigned)m;
        atomicMin(&keys[n], key);
    }
}

__global__ void gather_zq(const unsigned long long* __restrict__ keys,
                          const uint16_t* __restrict__ eb,
                          uint16_t* __restrict__ zq) {
    int n = blockIdx.x;
    unsigned idx = (unsigned)(keys[n] & 0xFFFFFFFFull);
    int b = n >> 8, hw = n & 255;
    for (int d = threadIdx.x; d < 512; d += blockDim.x)
        zq[((size_t)b * 512 + d) * 256 + hw] = eb[(size_t)idx * 512 + d];
}

// ---------------- host side ----------------

extern "C" void kernel_launch(void* const* d_in, const int* in_sizes, int n_in,
                              void* d_out, int out_size, void* d_ws, size_t ws_size,
                              hipStream_t stream) {
    (void)in_sizes; (void)n_in; (void)out_size; (void)ws_size;
    const float* x      = (const float*)d_in[0];
    const float* e_w[4] = {(const float*)d_in[1], (const float*)d_in[3],
                           (const float*)d_in[5], (const float*)d_in[7]};
    const float* e_b[4] = {(const float*)d_in[2], (const float*)d_in[4],
                           (const float*)d_in[6], (const float*)d_in[8]};
    const float* d_w[4] = {(const float*)d_in[9],  (const float*)d_in[11],
                           (const float*)d_in[13], (const float*)d_in[15]};
    const float* d_b[4] = {(const float*)d_in[10], (const float*)d_in[12],
                           (const float*)d_in[14], (const float*)d_in[16]};
    const float* embed  = (const float*)d_in[17];
    float* out = (float*)d_out;

    const int B = 32;
    const long ARENA = 33554432;  // u16 elems (fits largest activation: 32*64*128*128)

    char* p = (char*)d_ws;
    auto carve = [&](size_t bytes) {
        char* r = p;
        p += (bytes + 255) & ~(size_t)255;
        return r;
    };
    uint16_t* arena0 = (uint16_t*)carve(ARENA * 2);
    uint16_t* arena1 = (uint16_t*)carve(ARENA * 2);
    uint16_t* ew[4], * dw[4];
    ew[0] = (uint16_t*)carve(64L  * 48   * 2);
    ew[1] = (uint16_t*)carve(128L * 1024 * 2);
    ew[2] = (uint16_t*)carve(256L * 2048 * 2);
    ew[3] = (uint16_t*)carve(512L * 4096 * 2);
    dw[0] = (uint16_t*)carve(128L * 8192 * 2);
    dw[1] = (uint16_t*)carve(64L  * 2048 * 2);
    dw[2] = (uint16_t*)carve(32L  * 1024 * 2);
    dw[3] = (uint16_t*)carve(3L   * 512  * 2);
    uint16_t* embedb = (uint16_t*)carve(1024L * 512 * 2);
    float* cnorm     = (float*)carve(1024 * 4);
    unsigned long long* keys = (unsigned long long*)carve(8192 * 8);

    auto cast = [&](const float* src, uint16_t* dst, long n) {
        int blocks = (int)((n + 255) / 256);
        cast_f32_bf16<<<blocks, 256, 0, stream>>>(src, dst, n);
    };

    // weight / input preparation
    cast(x, arena0, (long)B * 3 * 256 * 256);
    cast(e_w[0], ew[0], 64L * 48);
    cast(e_w[1], ew[1], 128L * 1024);
    cast(e_w[2], ew[2], 256L * 2048);
    cast(e_w[3], ew[3], 512L * 4096);
    cast(embed, embedb, 1024L * 512);
    {
        int cins[4]  = {512, 128, 64, 32};
        int couts[4] = {128, 64, 32, 3};
        for (int i = 0; i < 4; ++i) {
            long total = (long)couts[i] * cins[i] * 16;
            permute_dec_w<<<(int)((total + 255) / 256), 256, 0, stream>>>(
                d_w[i], dw[i], cins[i], couts[i]);
        }
    }
    cnorm_kernel<<<4, 256, 0, stream>>>(embed, cnorm);
    init_keys<<<32, 256, 0, stream>>>(keys, 8192);

    auto conv = [&](const uint16_t* in, const uint16_t* w, const float* bias,
                    uint16_t* outb, float* outf,
                    int Cin, int Hin, int Win, int Cout, int Ho, int Wo,
                    int dilated, int ftanh) {
        long Ntot = (long)B * Ho * Wo;
        int  tm   = (Cout + 15) / 16;
        long tnw  = (Ntot + 63) / 64;           // 16x64 wave tiles along N
        long ngrp = (tnw + 7) / 8;              // 8 waves per block share weight tile
        int  blocks = (int)(tm * ngrp);
        int  K    = Cin * 16;
        int  Kpad = (K + 31) & ~31;
        int  KC   = Kpad < 1024 ? Kpad : 1024;
        size_t smem = (size_t)16 * KC * 2;      // weight tile in dynamic LDS
        conv_wmma_kernel<<<blocks, 256, smem, stream>>>(
            in, w, bias, outb, outf, B, Cin, Hin, Win, Cout, Ho, Wo, dilated, ftanh);
    };

    // encoder
    conv(arena0, ew[0], e_b[0], arena1, nullptr,   3, 256, 256,  64, 128, 128, 0, 0);
    conv(arena1, ew[1], e_b[1], arena0, nullptr,  64, 128, 128, 128,  64,  64, 0, 0);
    conv(arena0, ew[2], e_b[2], arena1, nullptr, 128,  64,  64, 256,  32,  32, 0, 0);
    conv(arena1, ew[3], e_b[3], arena0, nullptr, 256,  32,  32, 512,  16,  16, 0, 0);  // z

    // quantizer: 32768 wave-tiles -> 4096 blocks
    vq_wmma_kernel<<<4096, 256, 0, stream>>>(arena0, embedb, cnorm, keys);
    gather_zq<<<8192, 256, 0, stream>>>(keys, embedb, arena1);                          // z_q

    // decoder (transposed convs as dilated gathers)
    conv(arena1, dw[0], d_b[0], arena0, nullptr, 512,  16,  16, 128,  32,  32, 1, 0);
    conv(arena0, dw[1], d_b[1], arena1, nullptr, 128,  32,  32,  64,  64,  64, 1, 0);
    conv(arena1, dw[2], d_b[2], arena0, nullptr,  64,  64,  64,  32, 128, 128, 1, 0);
    conv(arena0, dw[3], d_b[3], nullptr, out,     32, 128, 128,   3, 256, 256, 1, 1);  // tanh, fp32
}
